// MultiHeadSelfAttentionModelV1_3951369913025
// MI455X (gfx1250) — matrix operand order repaired and checked by reading, not compiled, not run on hardware
//
#include <hip/hip_runtime.h>
#include <hip/hip_bf16.h>

// ---------------------------------------------------------------------------
// MHA forward for MI455X (gfx1250, wave32, WMMA).
// Precision: bf16 operands, fp32 accumulation via v_wmma_f32_16x16x32_bf16.
// Attention uses the 320KB WGP LDS to hold a full 16x2048 score strip
// (128KB fp32) + 16x2048 bf16 attn (64KB) -> exact 2-pass softmax in LDS.
// K is produced pre-transposed so the scores B-operand is contiguous.
// ---------------------------------------------------------------------------

typedef __bf16 bf16_t;
typedef __attribute__((ext_vector_type(16))) __bf16 v16bf;
typedef __attribute__((ext_vector_type(8)))  __bf16 v8bf;
typedef __attribute__((ext_vector_type(8)))  float  v8f;

#define BATCH 8
#define SEQ   2048
#define EMB   512
#define HEADS 8
#define DH    64
#define OUTC  10
#define NTOK  (BATCH * SEQ)   // 16384

__device__ __forceinline__ v16bf make_v16(v8bf lo, v8bf hi) {
  union { v16bf v; v8bf h[2]; } u;
  u.h[0] = lo; u.h[1] = hi;
  return u.v;
}

// A fragment (16 x 32, row-major, leading dim lda):
// lanes 0-15: row = lane,  K chunks {0-7} and {16-23}
// lanes16-31: row = lane-16, K chunks {8-15} and {24-31}
__device__ __forceinline__ v16bf load_a(const bf16_t* p, int row0, int k0, int lda) {
  int lane = threadIdx.x & 31;
  int r    = lane & 15;
  int half = lane >> 4;
  const bf16_t* base = p + (size_t)(row0 + r) * lda + k0 + half * 8;
  v8bf lo = *reinterpret_cast<const v8bf*>(base);
  v8bf hi = *reinterpret_cast<const v8bf*>(base + 16);
  return make_v16(lo, hi);
}

// B fragment (32 x 16, row-major, leading dim ldb):
// lane L holds column n0+(L&15), K rows {0-7},{16-23} (L<16) or {8-15},{24-31}.
__device__ __forceinline__ v16bf load_b(const bf16_t* p, int k0, int n0, int ldb) {
  int lane = threadIdx.x & 31;
  const bf16_t* base = p + (size_t)(k0 + (lane >> 4) * 8) * ldb + n0 + (lane & 15);
  v16bf v;
  for (int s = 0; s < 8; ++s) {
    v[s]     = base[(size_t)s * ldb];
    v[s + 8] = base[(size_t)(s + 16) * ldb];
  }
  return v;
}

__device__ __forceinline__ v8f wmma_bf16(v16bf a, v16bf b, v8f c) {
  // (neg_a, A, neg_b, B, c_mod, C, reuse_a, reuse_b)
  return __builtin_amdgcn_wmma_f32_16x16x32_bf16(false, a, false, b,
                                                 (short)0, c, false, false);
}

// ---------------------------------------------------------------------------
// K0: fp32 -> bf16 cast (weights)
// ---------------------------------------------------------------------------
__global__ void cvt_bf16_kernel(const float* __restrict__ src,
                                bf16_t* __restrict__ dst, int n) {
  int i = blockIdx.x * blockDim.x + threadIdx.x;
  if (i < n) dst[i] = (bf16_t)src[i];
}

// ---------------------------------------------------------------------------
// K1: embedding gather + cast, e_bf16[NTOK][EMB]
// ---------------------------------------------------------------------------
__global__ void embed_kernel(const int* __restrict__ x,
                             const float* __restrict__ table,
                             bf16_t* __restrict__ e) {
  int tok = blockIdx.x;
  int id  = x[tok];
  if (id < 0 || id >= 50257) id = 0;
  const float* row = table + (size_t)id * EMB;
  bf16_t*      dst = e     + (size_t)tok * EMB;
  for (int c = threadIdx.x; c < EMB; c += blockDim.x)
    dst[c] = (bf16_t)row[c];
}

// ---------------------------------------------------------------------------
// K2: fused QKV projection. grid = (NTOK/16, EMB/64, 3), block = 128 (4 waves)
// Q,V stored [B][H][S][DH]; K stored transposed [B][H][DH][S] so the
// attention scores B-operand (contraction over d) is contiguous.
// ---------------------------------------------------------------------------
__global__ __launch_bounds__(128)
void qkv_kernel(const bf16_t* __restrict__ e,
                const bf16_t* __restrict__ Wq, const bf16_t* __restrict__ Wk,
                const bf16_t* __restrict__ Wv,
                const float* __restrict__ bq, const float* __restrict__ bk,
                const float* __restrict__ bv,
                bf16_t* __restrict__ Q, bf16_t* __restrict__ Kt,
                bf16_t* __restrict__ V) {
  int mt    = blockIdx.x;      // 16-token row tile
  int g     = blockIdx.y;      // 64-wide column group
  int which = blockIdx.z;      // 0=Q, 1=K, 2=V
  int wave  = threadIdx.x >> 5;
  int n0    = g * 64 + wave * 16;

  const bf16_t* W    = (which == 0) ? Wq : (which == 1) ? Wk : Wv;
  const float*  bias = (which == 0) ? bq : (which == 1) ? bk : bv;

  v8f acc = {};
  for (int kc = 0; kc < EMB; kc += 32) {
    v16bf a = load_a(e, mt * 16, kc, EMB);
    v16bf b = load_b(W, kc, n0, EMB);
    acc = wmma_bf16(a, b, acc);
  }

  int lane = threadIdx.x & 31, r = lane & 15, half = lane >> 4;
  int col = n0 + r;
  int h = col >> 6, d = col & 63;
  float bcol = bias[col];
  for (int j = 0; j < 8; ++j) {
    int gm = mt * 16 + half * 8 + j;       // global token
    int b_ = gm >> 11, s = gm & (SEQ - 1);
    bf16_t val = (bf16_t)(acc[j] + bcol);
    if (which == 1)
      Kt[((size_t)(b_ * HEADS + h) * DH + d) * SEQ + s] = val;
    else {
      bf16_t* dst = (which == 0) ? Q : V;
      dst[((size_t)(b_ * HEADS + h) * SEQ + s) * DH + d] = val;
    }
  }
}

// ---------------------------------------------------------------------------
// K3: attention. grid = (SEQ/16, HEADS, BATCH), block = 128 (4 waves).
// LDS: fp32 scores[16][2048] (128KB) + bf16 attn[16][2048] (64KB) + red.
// ---------------------------------------------------------------------------
#define ATTN_LDS_BYTES (16 * SEQ * 4 + 16 * SEQ * 2 + (128 + 16) * 4)

__global__ __launch_bounds__(128)
void attn_kernel(const bf16_t* __restrict__ Q, const bf16_t* __restrict__ Kt,
                 const bf16_t* __restrict__ V, bf16_t* __restrict__ ctx) {
  extern __shared__ char smem[];
  float*  sc   = (float*)smem;                           // [16][SEQ]
  bf16_t* at   = (bf16_t*)(smem + 16 * SEQ * 4);         // [16][SEQ]
  float*  red  = (float*)(smem + 16 * SEQ * 4 + 16 * SEQ * 2); // [128]
  float*  rowv = red + 128;                              // [16]

  int qt = blockIdx.x, h = blockIdx.y, b = blockIdx.z;
  const bf16_t* Qb = Q  + (size_t)(b * HEADS + h) * SEQ * DH;
  const bf16_t* Kb = Kt + (size_t)(b * HEADS + h) * DH * SEQ;
  const bf16_t* Vb = V  + (size_t)(b * HEADS + h) * SEQ * DH;

  int wave = threadIdx.x >> 5, lane = threadIdx.x & 31;
  int r = lane & 15, half = lane >> 4;

  // ---- Phase 1: scores = Q K^T / sqrt(DH), into LDS ----
  v16bf a0 = load_a(Qb, qt * 16, 0,  DH);
  v16bf a1 = load_a(Qb, qt * 16, 32, DH);
  for (int kt = wave; kt < SEQ / 16; kt += 4) {
    v8f acc = {};
    acc = wmma_bf16(a0, load_b(Kb, 0,  kt * 16, SEQ), acc);
    acc = wmma_bf16(a1, load_b(Kb, 32, kt * 16, SEQ), acc);
    for (int j = 0; j < 8; ++j)
      sc[(half * 8 + j) * SEQ + kt * 16 + r] = acc[j] * 0.125f; // 1/sqrt(64)
  }
  __syncthreads();

  // ---- Phase 2: exact softmax in LDS (128 threads: 16 rows x 8 chunks) ----
  int row = threadIdx.x & 15, ch = threadIdx.x >> 4;
  int c0 = ch * (SEQ / 8), c1 = c0 + (SEQ / 8);
  float lm = -3.0e38f;
  for (int c = c0; c < c1; ++c) lm = fmaxf(lm, sc[row * SEQ + c]);
  red[row * 8 + ch] = lm;
  __syncthreads();
  if (threadIdx.x < 16) {
    float m = -3.0e38f;
    for (int i = 0; i < 8; ++i) m = fmaxf(m, red[threadIdx.x * 8 + i]);
    rowv[threadIdx.x] = m;
  }
  __syncthreads();
  float rmax = rowv[row];
  float ls = 0.0f;
  for (int c = c0; c < c1; ++c) ls += __expf(sc[row * SEQ + c] - rmax);
  red[row * 8 + ch] = ls;
  __syncthreads();
  if (threadIdx.x < 16) {
    float s = 0.0f;
    for (int i = 0; i < 8; ++i) s += red[threadIdx.x * 8 + i];
    rowv[threadIdx.x] = s;
  }
  __syncthreads();
  float inv = 1.0f / rowv[row];
  for (int c = c0; c < c1; ++c)
    at[row * SEQ + c] = (bf16_t)(__expf(sc[row * SEQ + c] - rmax) * inv);
  __syncthreads();

  // ---- Phase 3: ctx = attn @ V (A from LDS, B from global) ----
  int n0 = wave * 16;   // head-dim column tile (DH=64 -> 4 waves)
  v8f acc = {};
  for (int kc = 0; kc < SEQ; kc += 32) {
    v16bf a = load_a(at, 0, kc, SEQ);      // ds_load b128 fragments
    v16bf bb = load_b(Vb, kc, n0, DH);
    acc = wmma_bf16(a, bb, acc);
  }
  for (int j = 0; j < 8; ++j) {
    int s   = qt * 16 + half * 8 + j;
    int col = h * DH + n0 + r;
    ctx[((size_t)(b * SEQ + s)) * EMB + col] = (bf16_t)acc[j];
  }
}

// ---------------------------------------------------------------------------
// K4: out_vec = ctx @ Wo + bo, fused max-pool over sequence.
// grid = (EMB/16, BATCH), block = 128 (4 waves striding the 128 m-tiles).
// ---------------------------------------------------------------------------
__global__ __launch_bounds__(128)
void outproj_pool_kernel(const bf16_t* __restrict__ ctx,
                         const bf16_t* __restrict__ Wo,
                         const float* __restrict__ bo,
                         float* __restrict__ pooled) {
  __shared__ float red[4][16];
  int nt = blockIdx.x, b = blockIdx.y;
  int wave = threadIdx.x >> 5, lane = threadIdx.x & 31;
  int r = lane & 15, half = lane >> 4;
  int n0 = nt * 16;
  const bf16_t* cb = ctx + (size_t)b * SEQ * EMB;
  float bcol = bo[n0 + r];

  float vmax = -3.0e38f;
  for (int mt = wave; mt < SEQ / 16; mt += 4) {
    v8f acc = {};
    for (int kc = 0; kc < EMB; kc += 32) {
      v16bf a  = load_a(cb, mt * 16, kc, EMB);
      v16bf bb = load_b(Wo, kc, n0, EMB);
      acc = wmma_bf16(a, bb, acc);
    }
    for (int j = 0; j < 8; ++j) vmax = fmaxf(vmax, acc[j] + bcol);
  }
  // column n lives in lanes r and r+16 -> combine, then across 4 waves.
  vmax = fmaxf(vmax, __shfl_xor(vmax, 16));
  if (half == 0) red[wave][r] = vmax;
  __syncthreads();
  if (threadIdx.x < 16) {
    float m = red[0][threadIdx.x];
    for (int w = 1; w < 4; ++w) m = fmaxf(m, red[w][threadIdx.x]);
    pooled[b * EMB + n0 + threadIdx.x] = m;
  }
}

// ---------------------------------------------------------------------------
// K5: tiny classifier: out[b][o] = pooled[b] . Wc[:,o] + bc[o]  (8x10)
// ---------------------------------------------------------------------------
__global__ void classify_kernel(const float* __restrict__ pooled,
                                const float* __restrict__ Wc,
                                const float* __restrict__ bc,
                                float* __restrict__ out) {
  int t = threadIdx.x;
  if (t < BATCH * OUTC) {
    int b = t / OUTC, o = t % OUTC;
    float s = bc[o];
    for (int e = 0; e < EMB; ++e) s += pooled[b * EMB + e] * Wc[e * OUTC + o];
    out[b * OUTC + o] = s;
  }
}

// ---------------------------------------------------------------------------
// Host launcher
// ---------------------------------------------------------------------------
extern "C" void kernel_launch(void* const* d_in, const int* in_sizes, int n_in,
                              void* d_out, int out_size, void* d_ws, size_t ws_size,
                              hipStream_t stream) {
  const int*   x    = (const int*)d_in[0];
  const float* emb  = (const float*)d_in[1];
  const float* Wq   = (const float*)d_in[2];
  const float* bq   = (const float*)d_in[3];
  const float* Wk   = (const float*)d_in[4];
  const float* bk   = (const float*)d_in[5];
  const float* Wv   = (const float*)d_in[6];
  const float* bv   = (const float*)d_in[7];
  const float* Wo   = (const float*)d_in[8];
  const float* bo   = (const float*)d_in[9];
  const float* Wc   = (const float*)d_in[10];
  const float* bc   = (const float*)d_in[11];
  float* out = (float*)d_out;

  char* ws = (char*)d_ws;
  size_t off = 0;
  auto carve = [&](size_t bytes) {
    char* p = ws + off;
    off = (off + bytes + 255) & ~(size_t)255;
    return p;
  };
  bf16_t* eB   = (bf16_t*)carve((size_t)NTOK * EMB * 2);   // 16 MB
  bf16_t* WqB  = (bf16_t*)carve((size_t)EMB * EMB * 2);
  bf16_t* WkB  = (bf16_t*)carve((size_t)EMB * EMB * 2);
  bf16_t* WvB  = (bf16_t*)carve((size_t)EMB * EMB * 2);
  bf16_t* WoB  = (bf16_t*)carve((size_t)EMB * EMB * 2);
  bf16_t* Qb   = (bf16_t*)carve((size_t)NTOK * EMB * 2);   // [B][H][S][DH]
  bf16_t* Ktb  = (bf16_t*)carve((size_t)NTOK * EMB * 2);   // [B][H][DH][S]
  bf16_t* Vb   = (bf16_t*)carve((size_t)NTOK * EMB * 2);   // [B][H][S][DH]
  bf16_t* ctxB = (bf16_t*)carve((size_t)NTOK * EMB * 2);   // [B*S][EMB]
  float*  pooled = (float*)carve((size_t)BATCH * EMB * 4);
  (void)ws_size; (void)in_sizes; (void)n_in; (void)out_size;

  // K0: weight casts
  {
    int n = EMB * EMB, blk = 256, grd = (n + blk - 1) / blk;
    cvt_bf16_kernel<<<grd, blk, 0, stream>>>(Wq, WqB, n);
    cvt_bf16_kernel<<<grd, blk, 0, stream>>>(Wk, WkB, n);
    cvt_bf16_kernel<<<grd, blk, 0, stream>>>(Wv, WvB, n);
    cvt_bf16_kernel<<<grd, blk, 0, stream>>>(Wo, WoB, n);
  }
  // K1: embedding gather
  embed_kernel<<<NTOK, 128, 0, stream>>>(x, emb, eB);

  // K2: fused QKV projections (WMMA)
  qkv_kernel<<<dim3(NTOK / 16, EMB / 64, 3), 128, 0, stream>>>(
      eB, WqB, WkB, WvB, bq, bk, bv, Qb, Ktb, Vb);

  // K3: attention (scores WMMA -> LDS softmax -> ctx WMMA)
  attn_kernel<<<dim3(SEQ / 16, HEADS, BATCH), 128, ATTN_LDS_BYTES, stream>>>(
      Qb, Ktb, Vb, ctxB);

  // K4: output projection + max-pool
  outproj_pool_kernel<<<dim3(EMB / 16, BATCH), 128, 0, stream>>>(
      ctxB, WoB, bo, pooled);

  // K5: classifier
  classify_kernel<<<1, 128, 0, stream>>>(pooled, Wc, bc, out);
}
